// MSDeformAttnPixelDecoder_14851996910123
// MI455X (gfx1250) — compile-verified, matrix operand order checked by reading, not run on hardware
//
#include <hip/hip_runtime.h>

#define LEN   5376

typedef __attribute__((ext_vector_type(16))) __bf16 v16bf;
typedef __attribute__((ext_vector_type(8)))  float  v8f;

union BFTile { v16bf v; uint4 q[2]; unsigned short u[16]; };

__device__ __forceinline__ unsigned short f2bf(float f) {
  union { float f; unsigned u; } x; x.f = f;
  unsigned u = x.u;
  return (unsigned short)((u + 0x7FFFu + ((u >> 16) & 1u)) >> 16);
}

// bf16-resident 16-row x 32-K tile in CDNA5 16-bit A/B VGPR layout.
// lane l (r=l&15, hi=l>>4): elements 0..7 = K[hi*8 .. hi*8+7],
// elements 8..15 = K[16+hi*8 .. 23+hi*8]  -> two 16B vector loads.
__device__ __forceinline__ v16bf load_tile_bf(const unsigned short* __restrict__ base,
                                              long rowStride, int lane) {
  int r = lane & 15, hi = lane >> 4;
  const unsigned short* p = base + (long)r * rowStride + hi * 8;
  BFTile t;
  t.q[0] = *(const uint4*)p;
  t.q[1] = *(const uint4*)(p + 16);
  return t.v;
}

// ---------------------------------------------------------------------------
// Linear: C[M,N] = A[M,K]*W[N,K]^T + bias ; A,W bf16 row-major, C f32 or bf16
// Each wave computes a 32x(16*NT) output tile: W tiles reused for 2 M-tiles.
// ---------------------------------------------------------------------------
template<int NT, bool RELU, bool ACC, bool OBF>
__global__ __launch_bounds__(256) void gemm_bf16(
    const unsigned short* __restrict__ A, const unsigned short* __restrict__ W,
    const float* __restrict__ bias, void* __restrict__ Cv,
    int M, int N, int K, long aBatch, long cBatch) {
  const int lane = threadIdx.x & 31;
  const int wave = (int)(((long)blockIdx.x * blockDim.x + threadIdx.x) >> 5);
  const int NG = N / (16 * NT);
  const int Mt = M / 32;
  if (wave >= Mt * NG) return;
  const int mt = wave / NG;
  const int ng = wave % NG;
  const int b  = blockIdx.z;
  const unsigned short* Ab0 = A + (long)b * aBatch + (long)(mt * 32) * K;
  const unsigned short* Ab1 = Ab0 + (long)16 * K;
  const unsigned short* W0  = W + (long)(ng * NT * 16) * K;

  v8f acc[2][NT] = {};
  for (int k0 = 0; k0 < K; k0 += 32) {
    v16bf a0 = load_tile_bf(Ab0 + k0, K, lane);
    v16bf a1 = load_tile_bf(Ab1 + k0, K, lane);
#pragma unroll
    for (int t = 0; t < NT; ++t) {
      v16bf bm = load_tile_bf(W0 + (long)(t * 16) * K + k0, K, lane);
      acc[0][t] = __builtin_amdgcn_wmma_f32_16x16x32_bf16(false, a0, false, bm,
                                                          (short)0, acc[0][t], false, false);
      acc[1][t] = __builtin_amdgcn_wmma_f32_16x16x32_bf16(false, a1, false, bm,
                                                          (short)0, acc[1][t], false, false);
    }
  }
  const int r0 = (lane >> 4) * 8;
  const int nc = lane & 15;
  const float* Cf = (const float*)Cv;
#pragma unroll
  for (int mi = 0; mi < 2; ++mi) {
#pragma unroll
    for (int t = 0; t < NT; ++t) {
      int n = ng * NT * 16 + t * 16 + nc;
      float bv = bias ? bias[n] : 0.f;
#pragma unroll
      for (int r = 0; r < 8; ++r) {
        int m = mt * 32 + mi * 16 + r0 + r;
        long ci = (long)b * cBatch + (long)m * N + n;
        float v = acc[mi][t][r] + bv;
        if (ACC && !OBF) v += Cf[ci];
        if (RELU) v = v > 0.f ? v : 0.f;
        if (OBF) ((unsigned short*)Cv)[ci] = f2bf(v);
        else     ((float*)Cv)[ci] = v;
      }
    }
  }
}

// ---------------------------------------------------------------------------
// 3x3 conv tap: token-major bf16 input (B,16384,256), repacked bf16 weights
// wave computes 32x64 tile (two independently-shifted 16-pixel row tiles)
// ---------------------------------------------------------------------------
template<bool ACC>
__global__ __launch_bounds__(256) void conv3x3_tap(
    const unsigned short* __restrict__ T, const unsigned short* __restrict__ W,
    float* __restrict__ C, int dy, int dx) {
  const int lane = threadIdx.x & 31;
  const int wave = (int)(((long)blockIdx.x * blockDim.x + threadIdx.x) >> 5);
  const int Mt = 16384 / 32;
  if (wave >= Mt * 4) return;
  const int mt = wave >> 2;
  const int ng = wave & 3;
  const int b  = blockIdx.z;
  const unsigned short* Tb = T + (long)b * 16384 * 256;
  float* Cb = C + (long)b * 16384 * 256;

  int r = lane & 15, hi = lane >> 4;
  bool valid[2];
  const unsigned short* arow[2];
#pragma unroll
  for (int mi = 0; mi < 2; ++mi) {
    int p = mt * 32 + mi * 16 + r;
    int y = p >> 7, x = p & 127;
    int y2 = y + dy, x2 = x + dx;
    valid[mi] = (y2 >= 0 && y2 < 128 && x2 >= 0 && x2 < 128);
    arow[mi] = Tb + (long)(y2 * 128 + x2) * 256 + hi * 8;
  }

  v8f acc[2][4] = {};
  for (int k0 = 0; k0 < 256; k0 += 32) {
    BFTile ta[2];
#pragma unroll
    for (int mi = 0; mi < 2; ++mi) {
      if (valid[mi]) {
        ta[mi].q[0] = *(const uint4*)(arow[mi] + k0);
        ta[mi].q[1] = *(const uint4*)(arow[mi] + k0 + 16);
      } else {
#pragma unroll
        for (int j = 0; j < 16; ++j) ta[mi].u[j] = 0;
      }
    }
#pragma unroll
    for (int t = 0; t < 4; ++t) {
      v16bf bm = load_tile_bf(W + (long)(ng * 64 + t * 16) * 256 + k0, 256, lane);
      acc[0][t] = __builtin_amdgcn_wmma_f32_16x16x32_bf16(false, ta[0].v, false, bm,
                                                          (short)0, acc[0][t], false, false);
      acc[1][t] = __builtin_amdgcn_wmma_f32_16x16x32_bf16(false, ta[1].v, false, bm,
                                                          (short)0, acc[1][t], false, false);
    }
  }
  const int r0 = (lane >> 4) * 8;
  const int nc = lane & 15;
#pragma unroll
  for (int mi = 0; mi < 2; ++mi) {
#pragma unroll
    for (int t = 0; t < 4; ++t) {
      int n = ng * 64 + t * 16 + nc;
#pragma unroll
      for (int q = 0; q < 8; ++q) {
        long ci = (long)(mt * 32 + mi * 16 + r0 + q) * 256 + n;
        float v = acc[mi][t][q];
        if (ACC) v += Cb[ci];
        Cb[ci] = v;
      }
    }
  }
}

// ------------------------- conversion / repack -----------------------------
__global__ void cvt_bf16(const float* __restrict__ s, unsigned short* __restrict__ d, long n) {
  long i = (long)blockIdx.x * 256 + threadIdx.x;
  if (i < n) d[i] = f2bf(s[i]);
}

// NCHW f32 -> token-major bf16 (B,HW,C)
__global__ void chw_to_tok_bf16(const float* __restrict__ s, unsigned short* __restrict__ d,
                                int HW, int C, int B) {
  long idx = (long)blockIdx.x * 256 + threadIdx.x;
  long total = (long)B * HW * C;
  if (idx >= total) return;
  int c = (int)(idx % C);
  long t2 = idx / C;
  int p = (int)(t2 % HW);
  int b = (int)(t2 / HW);
  d[idx] = f2bf(s[((long)b * C + c) * HW + p]);
}

// outcw (O,I,3,3) -> 9 K-major bf16 matrices: d[(tap*256+o)*256+c]
__global__ void repack_conv_w(const float* __restrict__ w, unsigned short* __restrict__ d) {
  long idx = (long)blockIdx.x * 256 + threadIdx.x;
  if (idx >= 9L * 256 * 256) return;
  int c = (int)(idx & 255);
  long t = idx >> 8;
  int o = (int)(t & 255);
  int tap = (int)(t >> 8);
  int ky = tap / 3, kx = tap % 3;
  d[idx] = f2bf(w[(((long)o * 256 + c) * 3 + ky) * 3 + kx]);
}

// ----------------------------- GroupNorm -----------------------------------
__global__ __launch_bounds__(256) void gn_stats(const float* __restrict__ s,
                                                float* __restrict__ stats, int HW) {
  int bg = blockIdx.x;
  int b = bg >> 5, g = bg & 31;
  long n = (long)HW * 8;
  float sum = 0.f, sq = 0.f;
  for (long i = threadIdx.x; i < n; i += 256) {
    int p = (int)(i >> 3), j = (int)(i & 7);
    float v = s[((long)b * HW + p) * 256 + g * 8 + j];
    sum += v; sq += v * v;
  }
  for (int o = 16; o > 0; o >>= 1) { sum += __shfl_xor(sum, o, 32); sq += __shfl_xor(sq, o, 32); }
  __shared__ float ss[8], s2[8];
  int w = threadIdx.x >> 5, lane = threadIdx.x & 31;
  if (lane == 0) { ss[w] = sum; s2[w] = sq; }
  __syncthreads();
  if (threadIdx.x == 0) {
    float S = 0.f, Q = 0.f;
    for (int i = 0; i < 8; ++i) { S += ss[i]; Q += s2[i]; }
    float m = S / (float)n;
    stats[bg * 2] = m;
    stats[bg * 2 + 1] = rsqrtf(Q / (float)n - m * m + 1e-5f);
  }
}

__global__ void gn_apply(const float* __restrict__ s, float* __restrict__ dst,
                         unsigned short* __restrict__ dstbf,
                         const float* __restrict__ stats,
                         const float* __restrict__ g, const float* __restrict__ bt,
                         int HW, long sBatch, long dBatch, int relu, int B) {
  long idx = (long)blockIdx.x * 256 + threadIdx.x;
  long total = (long)B * HW * 256;
  if (idx >= total) return;
  int c = (int)(idx & 255);
  long pbp = idx >> 8;
  int p = (int)(pbp % HW);
  int b = (int)(pbp / HW);
  const float* st = stats + ((long)b * 32 + (c >> 3)) * 2;
  float v = (s[(long)b * sBatch + (long)p * 256 + c] - st[0]) * st[1] * g[c] + bt[c];
  if (relu) v = v > 0.f ? v : 0.f;
  long di = (long)b * dBatch + (long)p * 256 + c;
  dst[di] = v;
  if (dstbf) dstbf[di] = f2bf(v);
}

// ----------------- residual + LayerNorm (dual fp32/bf16) -------------------
__global__ __launch_bounds__(256) void add_ln(float* __restrict__ src,
                                              unsigned short* __restrict__ srcbf,
                                              const float* __restrict__ delta,
                                              const float* __restrict__ g,
                                              const float* __restrict__ bt, int Mtok) {
  int wave = (int)(((long)blockIdx.x * blockDim.x + threadIdx.x) >> 5);
  int lane = threadIdx.x & 31;
  if (wave >= Mtok) return;
  long base = (long)wave * 256 + lane * 8;
  float v[8]; float s = 0.f, q = 0.f;
#pragma unroll
  for (int j = 0; j < 8; ++j) { v[j] = src[base + j] + delta[base + j]; s += v[j]; q += v[j] * v[j]; }
  for (int o = 16; o > 0; o >>= 1) { s += __shfl_xor(s, o, 32); q += __shfl_xor(q, o, 32); }
  float m = s * (1.f / 256.f);
  float rstd = rsqrtf(q * (1.f / 256.f) - m * m + 1e-5f);
  int c0 = lane * 8;
#pragma unroll
  for (int j = 0; j < 8; ++j) {
    float o2 = (v[j] - m) * rstd * g[c0 + j] + bt[c0 + j];
    src[base + j] = o2;
    srcbf[base + j] = f2bf(o2);
  }
}

// --------------------- pos embed / q / softmax -----------------------------
__global__ void pos_embed_kernel(float* __restrict__ pos, const float* __restrict__ lvl) {
  long idx = (long)blockIdx.x * 256 + threadIdx.x;
  if (idx >= (long)LEN * 256) return;
  int c = (int)(idx & 255);
  int t = (int)(idx >> 8);
  int l, S, p;
  if (t < 256) { l = 0; S = 16; p = t; }
  else if (t < 1280) { l = 1; S = 32; p = t - 256; }
  else { l = 2; S = 64; p = t - 1280; }
  int y = p / S, x = p % S;
  const float TWO_PI = 6.283185307179586f;
  float denom = (float)S + 1e-6f;
  float v; int cc = c;
  if (c < 128) v = (float)(y + 1) / denom * TWO_PI;
  else { v = (float)(x + 1) / denom * TWO_PI; cc = c - 128; }
  float d = powf(10000.f, (float)(cc & ~1) / 128.f);
  float a = v / d;
  pos[idx] = ((cc & 1) ? cosf(a) : sinf(a)) + lvl[l * 256 + c];
}

__global__ void q_add(unsigned short* __restrict__ qbf, const float* __restrict__ src,
                      const float* __restrict__ pos, long total) {
  long idx = (long)blockIdx.x * 256 + threadIdx.x;
  if (idx >= total) return;
  qbf[idx] = f2bf(src[idx] + pos[idx % ((long)LEN * 256)]);
}

__global__ void softmax12(float* __restrict__ aw, int total) {
  int i = (int)((long)blockIdx.x * 256 + threadIdx.x);
  if (i >= total) return;
  float* p = aw + (long)(i / 8) * 96 + (i % 8) * 12;
  float mx = p[0];
#pragma unroll
  for (int j = 1; j < 12; ++j) mx = fmaxf(mx, p[j]);
  float e[12]; float s = 0.f;
#pragma unroll
  for (int j = 0; j < 12; ++j) { e[j] = expf(p[j] - mx); s += e[j]; }
  float inv = 1.f / s;
#pragma unroll
  for (int j = 0; j < 12; ++j) p[j] = e[j] * inv;
}

// --------------------- deformable sampling (bf16 out) ----------------------
__global__ __launch_bounds__(256) void deform_attn(const float* __restrict__ valb,
                                                   const float* __restrict__ offb,
                                                   const float* __restrict__ awb,
                                                   unsigned short* __restrict__ attn_sbf, int B) {
  int gw = (int)(((long)blockIdx.x * blockDim.x + threadIdx.x) >> 5);
  int lane = threadIdx.x & 31;
  int total = B * LEN * 8;
  if (gw >= total) return;
  int h = gw & 7;
  int bt = gw >> 3;
  int t = bt % LEN;
  int b = bt / LEN;
  int S0, p0;
  if (t < 256) { S0 = 16; p0 = t; }
  else if (t < 1280) { S0 = 32; p0 = t - 256; }
  else { S0 = 64; p0 = t - 1280; }
  float refx = ((p0 % S0) + 0.5f) / (float)S0;
  float refy = ((p0 / S0) + 0.5f) / (float)S0;
  const float* off = offb + (long)bt * 192 + h * 24;
  const float* aw  = awb  + (long)bt * 96  + h * 12;
  long vbase = ((long)b * LEN) * 256 + h * 32 + lane;
  const int lvlOff[3] = {0, 256, 1280};
  const int lvlS[3]   = {16, 32, 64};
  float acc = 0.f;
#pragma unroll
  for (int l = 0; l < 3; ++l) {
    int S = lvlS[l], O = lvlOff[l];
    float inv = 1.f / (float)S;
#pragma unroll
    for (int pt = 0; pt < 4; ++pt) {
      float ox = off[l * 8 + pt * 2 + 0];
      float oy = off[l * 8 + pt * 2 + 1];
      float w  = aw[l * 4 + pt];
      float xx = (refx + ox * inv) * S - 0.5f;
      float yy = (refy + oy * inv) * S - 0.5f;
      float xf = floorf(xx), yf = floorf(yy);
      int x0 = (int)xf, y0 = (int)yf;
      float wx = xx - xf, wy = yy - yf;
      float v00 = 0.f, v01 = 0.f, v10 = 0.f, v11 = 0.f;
      if (y0 >= 0 && y0 < S) {
        if (x0 >= 0 && x0 < S)         v00 = valb[vbase + (long)(O + y0 * S + x0) * 256];
        if (x0 + 1 >= 0 && x0 + 1 < S) v01 = valb[vbase + (long)(O + y0 * S + x0 + 1) * 256];
      }
      if (y0 + 1 >= 0 && y0 + 1 < S) {
        if (x0 >= 0 && x0 < S)         v10 = valb[vbase + (long)(O + (y0 + 1) * S + x0) * 256];
        if (x0 + 1 >= 0 && x0 + 1 < S) v11 = valb[vbase + (long)(O + (y0 + 1) * S + x0 + 1) * 256];
      }
      acc += w * ((1.f - wy) * ((1.f - wx) * v00 + wx * v01)
                +        wy  * ((1.f - wx) * v10 + wx * v11));
    }
  }
  attn_sbf[(long)bt * 256 + h * 32 + lane] = f2bf(acc);
}

// bilinear 64->128 upsample of src level-2 slice + cur -> bf16 conv input
__global__ void upsample_add(const float* __restrict__ cur, const float* __restrict__ src,
                             unsigned short* __restrict__ out_bf, int B) {
  long idx = (long)blockIdx.x * 256 + threadIdx.x;
  long total = (long)B * 16384 * 256;
  if (idx >= total) return;
  int c = (int)(idx & 255);
  long pb = idx >> 8;
  int p = (int)(pb % 16384);
  int b = (int)(pb / 16384);
  int y = p >> 7, x = p & 127;
  float sy = (y + 0.5f) * 0.5f - 0.5f;
  float sx = (x + 0.5f) * 0.5f - 0.5f;
  float fy = floorf(sy), fx = floorf(sx);
  float wy = sy - fy, wx = sx - fx;
  int y0 = (int)fy, x0 = (int)fx;
  int y0c = min(63, max(0, y0)), y1c = min(63, max(0, y0 + 1));
  int x0c = min(63, max(0, x0)), x1c = min(63, max(0, x0 + 1));
  long sb = ((long)b * LEN + 1280) * 256 + c;
  float v00 = src[sb + (long)(y0c * 64 + x0c) * 256];
  float v01 = src[sb + (long)(y0c * 64 + x1c) * 256];
  float v10 = src[sb + (long)(y1c * 64 + x0c) * 256];
  float v11 = src[sb + (long)(y1c * 64 + x1c) * 256];
  float up = (1.f - wy) * ((1.f - wx) * v00 + wx * v01)
           +        wy  * ((1.f - wx) * v10 + wx * v11);
  out_bf[idx] = f2bf(cur[idx] + up);
}

// token-major f32 (B,HW,256) -> NCHW f32
__global__ void to_chw(const float* __restrict__ s, float* __restrict__ dst,
                       int HW, long sBatch, long dBatch, int B) {
  long idx = (long)blockIdx.x * 256 + threadIdx.x;
  long total = (long)B * 256 * HW;
  if (idx >= total) return;
  int p = (int)(idx % HW);
  long t2 = idx / HW;
  int c = (int)(t2 % 256);
  int b = (int)(t2 / 256);
  dst[(long)b * dBatch + (long)c * HW + p] = s[(long)b * sBatch + (long)p * 256 + c];
}

// ===========================================================================
extern "C" void kernel_launch(void* const* d_in, const int* in_sizes, int n_in,
                              void* d_out, int out_size, void* d_ws, size_t ws_size,
                              hipStream_t stream) {
  (void)n_in; (void)out_size; (void)ws_size;
  const float* res2 = (const float*)d_in[0];
  const float* res3 = (const float*)d_in[1];
  const float* res4 = (const float*)d_in[2];
  const float* res5 = (const float*)d_in[3];
  const float* pw5  = (const float*)d_in[4];  const float* pb5 = (const float*)d_in[5];
  const float* pw4  = (const float*)d_in[6];  const float* pb4 = (const float*)d_in[7];
  const float* pw3  = (const float*)d_in[8];  const float* pb3 = (const float*)d_in[9];
  const float* gn_g = (const float*)d_in[10]; const float* gn_b = (const float*)d_in[11];
  const float* lemb = (const float*)d_in[12];
  const float* off_w = (const float*)d_in[13]; const float* off_b = (const float*)d_in[14];
  const float* aw_w  = (const float*)d_in[15]; const float* aw_b  = (const float*)d_in[16];
  const float* val_w = (const float*)d_in[17]; const float* val_b = (const float*)d_in[18];
  const float* ow    = (const float*)d_in[19]; const float* ob    = (const float*)d_in[20];
  const float* ln1g  = (const float*)d_in[21]; const float* ln1b  = (const float*)d_in[22];
  const float* f1w   = (const float*)d_in[23]; const float* f1b   = (const float*)d_in[24];
  const float* f2w   = (const float*)d_in[25]; const float* f2b   = (const float*)d_in[26];
  const float* ln2g  = (const float*)d_in[27]; const float* ln2b  = (const float*)d_in[28];
  const float* lat_w = (const float*)d_in[29];
  const float* latg  = (const float*)d_in[30]; const float* latb  = (const float*)d_in[31];
  const float* outcw = (const float*)d_in[32];
  const float* outcg = (const float*)d_in[33]; const float* outcb = (const float*)d_in[34];
  const float* mfw   = (const float*)d_in[35]; const float* mfb   = (const float*)d_in[36];

  const int B = in_sizes[0] / (256 * 128 * 128);
  const long Mtok = (long)B * LEN;
  const long SRC_SZ = Mtok * 256;

  // ---- fp32 workspace ----
  float* ws    = (float*)d_ws;
  float* src   = ws;                          // Mtok*256
  float* pos   = src + SRC_SZ;                // LEN*256
  float* stats = pos + (long)LEN * 256;       // 4096
  float* pool  = stats + 4096;                // 25,165,824 f32 (FPN dominates)
  float* valb   = pool;                       // Mtok*256
  float* offb   = valb + SRC_SZ;              // Mtok*192
  float* awb    = offb + Mtok * 192;          // Mtok*96
  float* attn_o = awb + Mtok * 96;            // Mtok*256
  const long FPN_SZ = (long)B * 16384 * 256;
  float* fpnA = pool;
  float* fpnC = pool + FPN_SZ;
  float* mfT  = pool + 2 * FPN_SZ;

  // ---- bf16 workspace ----
  unsigned short* bfpool = (unsigned short*)(pool + 3 * FPN_SZ);  // 25,165,824 ush
  unsigned short* srcbf   = bfpool;                 // Mtok*256
  unsigned short* qbf     = srcbf + SRC_SZ;         // Mtok*256
  unsigned short* attn_sbf= qbf + SRC_SZ;           // Mtok*256
  unsigned short* ffn_hbf = attn_sbf + SRC_SZ;      // Mtok*1024
  unsigned short* lvlbf   = ffn_hbf;                // setup-phase alias
  unsigned short* r2t     = bfpool;                 // FPN-phase aliases
  unsigned short* fpnAbf  = bfpool + FPN_SZ;
  unsigned short* fpnCbf  = bfpool + 2 * FPN_SZ;

  unsigned short* wbf = bfpool + 3 * FPN_SZ;
  unsigned short* wpw5 = wbf;               // 524288
  unsigned short* wpw4 = wpw5 + 524288;     // 262144
  unsigned short* wpw3 = wpw4 + 262144;     // 131072
  unsigned short* woff = wpw3 + 131072;     // 294912
  unsigned short* waw  = woff + 294912;     // 147456
  unsigned short* wval = waw + 147456;      // 393216
  unsigned short* wow  = wval + 393216;     // 393216
  unsigned short* wf1  = wow + 393216;      // 1572864
  unsigned short* wf2  = wf1 + 1572864;     // 1572864
  unsigned short* wlat = wf2 + 1572864;     // 65536
  unsigned short* wcv  = wlat + 65536;      // 589824
  unsigned short* wmf  = wcv + 589824;      // 65536

  auto blk = [](long n) { return (unsigned)((n + 255) / 256); };
  auto wvb = [](long waves) { return (unsigned)((waves + 7) / 8); };
  auto cvt = [&](const float* s, unsigned short* d, long n) {
    cvt_bf16<<<blk(n), 256, 0, stream>>>(s, d, n);
  };

  // ---- one-time weight conversion ----
  cvt(pw5, wpw5, 524288);  cvt(pw4, wpw4, 262144);  cvt(pw3, wpw3, 131072);
  cvt(off_w, woff, 294912); cvt(aw_w, waw, 147456);
  cvt(val_w, wval, 393216); cvt(ow, wow, 393216);
  cvt(f1w, wf1, 1572864);   cvt(f2w, wf2, 1572864);
  cvt(lat_w, wlat, 65536);  cvt(mfw, wmf, 65536);
  repack_conv_w<<<blk(9L * 65536), 256, 0, stream>>>(outcw, wcv);

  // ---- input projections + GroupNorm -> src (+ srcbf) ----
  struct Lv { const float* f; const unsigned short* w; const float* bias; int HW; int K; int off; };
  Lv lv[3] = { { res5, wpw5, pb5, 256, 2048, 0 },
               { res4, wpw4, pb4, 1024, 1024, 256 },
               { res3, wpw3, pb3, 4096, 512, 1280 } };
  for (int i = 0; i < 3; ++i) {
    int M = lv[i].HW, K = lv[i].K;
    chw_to_tok_bf16<<<blk((long)B * M * K), 256, 0, stream>>>(lv[i].f, lvlbf, M, K, B);
    long waves = (long)(M / 32) * 4;
    gemm_bf16<4, false, false, false><<<dim3(wvb(waves), 1, B), 256, 0, stream>>>(
        lvlbf, lv[i].w, lv[i].bias, pool, M, 256, K, (long)M * K, (long)M * 256);
    gn_stats<<<B * 32, 256, 0, stream>>>(pool, stats, M);
    gn_apply<<<blk((long)B * M * 256), 256, 0, stream>>>(
        pool, src + (long)lv[i].off * 256, srcbf + (long)lv[i].off * 256, stats,
        gn_g + i * 256, gn_b + i * 256, M, (long)M * 256, (long)LEN * 256, 0, B);
  }

  pos_embed_kernel<<<(LEN * 256) / 256, 256, 0, stream>>>(pos, lemb);

  // ---- 6 encoder layers ----
  long w256 = (Mtok / 32) * 4;
  for (int i = 0; i < 6; ++i) {
    q_add<<<blk(Mtok * 256), 256, 0, stream>>>(qbf, src, pos, Mtok * 256);

    gemm_bf16<4, false, false, false><<<dim3(wvb((Mtok / 32) * 3), 1, 1), 256, 0, stream>>>(
        qbf, woff + (long)i * 49152, off_b + i * 192, offb, (int)Mtok, 192, 256, 0, 0);
    gemm_bf16<2, false, false, false><<<dim3(wvb((Mtok / 32) * 3), 1, 1), 256, 0, stream>>>(
        qbf, waw + (long)i * 24576, aw_b + i * 96, awb, (int)Mtok, 96, 256, 0, 0);
    softmax12<<<blk(Mtok * 8), 256, 0, stream>>>(awb, (int)(Mtok * 8));

    gemm_bf16<4, false, false, false><<<dim3(wvb(w256), 1, 1), 256, 0, stream>>>(
        srcbf, wval + (long)i * 65536, val_b + i * 256, valb, (int)Mtok, 256, 256, 0, 0);

    deform_attn<<<wvb(Mtok * 8), 256, 0, stream>>>(valb, offb, awb, attn_sbf, B);

    gemm_bf16<4, false, false, false><<<dim3(wvb(w256), 1, 1), 256, 0, stream>>>(
        attn_sbf, wow + (long)i * 65536, ob + i * 256, attn_o, (int)Mtok, 256, 256, 0, 0);
    add_ln<<<wvb(Mtok), 256, 0, stream>>>(src, srcbf, attn_o,
                                          ln1g + i * 256, ln1b + i * 256, (int)Mtok);

    gemm_bf16<4, true, false, true><<<dim3(wvb((Mtok / 32) * 16), 1, 1), 256, 0, stream>>>(
        srcbf, wf1 + (long)i * 262144, f1b + i * 1024, ffn_hbf, (int)Mtok, 1024, 256, 0, 0);
    gemm_bf16<4, false, false, false><<<dim3(wvb(w256), 1, 1), 256, 0, stream>>>(
        ffn_hbf, wf2 + (long)i * 262144, f2b + i * 256, attn_o, (int)Mtok, 256, 1024, 0, 0);
    add_ln<<<wvb(Mtok), 256, 0, stream>>>(src, srcbf, attn_o,
                                          ln2g + i * 256, ln2b + i * 256, (int)Mtok);
  }

  // ---- multi-scale outputs ----
  float* out_mask = (float*)d_out;
  float* out0 = out_mask + (long)B * 256 * 16384;
  float* out1 = out0 + (long)B * 256 * 256;
  float* out2 = out1 + (long)B * 256 * 1024;
  to_chw<<<blk((long)B * 256 * 256), 256, 0, stream>>>(src, out0, 256, (long)LEN * 256, 256 * 256, B);
  to_chw<<<blk((long)B * 256 * 1024), 256, 0, stream>>>(src + 256 * 256, out1, 1024, (long)LEN * 256, 256 * 1024, B);
  to_chw<<<blk((long)B * 256 * 4096), 256, 0, stream>>>(src + 1280 * 256, out2, 4096, (long)LEN * 256, 256 * 4096, B);

  // ---- FPN ----
  long ftot = (long)B * 16384 * 256;
  long fwaves = (16384 / 32) * 4;
  chw_to_tok_bf16<<<blk(ftot), 256, 0, stream>>>(res2, r2t, 16384, 256, B);
  gemm_bf16<4, false, false, false><<<dim3(wvb(fwaves), 1, B), 256, 0, stream>>>(
      r2t, wlat, nullptr, fpnA, 16384, 256, 256, (long)16384 * 256, (long)16384 * 256);
  gn_stats<<<B * 32, 256, 0, stream>>>(fpnA, stats, 16384);
  gn_apply<<<blk(ftot), 256, 0, stream>>>(
      fpnA, fpnA, (unsigned short*)nullptr, stats, latg, latb,
      16384, (long)16384 * 256, (long)16384 * 256, 0, B);
  upsample_add<<<blk(ftot), 256, 0, stream>>>(fpnA, src, fpnAbf, B);

  for (int ky = 0; ky < 3; ++ky)
    for (int kx = 0; kx < 3; ++kx) {
      int tap = ky * 3 + kx;
      if (tap == 0)
        conv3x3_tap<false><<<dim3(wvb(fwaves), 1, B), 256, 0, stream>>>(
            fpnAbf, wcv + (long)tap * 65536, fpnC, ky - 1, kx - 1);
      else
        conv3x3_tap<true><<<dim3(wvb(fwaves), 1, B), 256, 0, stream>>>(
            fpnAbf, wcv + (long)tap * 65536, fpnC, ky - 1, kx - 1);
    }
  gn_stats<<<B * 32, 256, 0, stream>>>(fpnC, stats, 16384);
  gn_apply<<<blk(ftot), 256, 0, stream>>>(
      fpnC, fpnC, fpnCbf, stats, outcg, outcb,
      16384, (long)16384 * 256, (long)16384 * 256, 1, B);

  gemm_bf16<4, false, false, false><<<dim3(wvb(fwaves), 1, B), 256, 0, stream>>>(
      fpnCbf, wmf, mfb, mfT, 16384, 256, 256, (long)16384 * 256, (long)16384 * 256);
  to_chw<<<blk(ftot), 256, 0, stream>>>(
      mfT, out_mask, 16384, (long)16384 * 256, (long)256 * 16384, B);
}
